// LinearAttention_29609504539340
// MI455X (gfx1250) — matrix operand order repaired and checked by reading, not compiled, not run on hardware
//
#include <hip/hip_runtime.h>
#include <cstdint>
#include <cstddef>

// ---------------------------------------------------------------------------
// MI455X (gfx1250): bf16 WMMA (16x16x32, f32 accum) GEMMs, f32 VALU epilogues.
// Global->LDS staging now uses CDNA5 async copies (GLOBAL_LOAD_ASYNC_TO_LDS)
// tracked by ASYNCcnt, so tile kt+1 streams into LDS while tile kt runs WMMA.
// ---------------------------------------------------------------------------

typedef __attribute__((ext_vector_type(16))) __bf16 bf16x16;
typedef __attribute__((ext_vector_type(8)))  __bf16 bf16x8;
typedef __attribute__((ext_vector_type(8)))  float  f32x8;

#define BM 128
#define BN 128
#define BK 32
#define LDSS 40   // padded K-stride (bf16 elems) = 80B rows -> conflict-free-ish

enum { EPI_BIAS = 0, EPI_SIGMOID, EPI_ELU1, EPI_BF16, EPI_SCALE, EPI_MULY, EPI_BIASRES };

static __device__ __forceinline__ bf16x16 cat8(bf16x8 lo, bf16x8 hi) {
  return __builtin_shufflevector(lo, hi, 0,1,2,3,4,5,6,7,8,9,10,11,12,13,14,15);
}

// 32B global->LDS async copy (two b128 ops; INST_OFFSET applies to both sides)
static __device__ __forceinline__ void async_copy32(void* lds_dst, const void* gsrc) {
  const unsigned dst = (unsigned)(uintptr_t)lds_dst;   // wave-relative LDS byte addr
  asm volatile(
      "global_load_async_to_lds_b128 %0, %1, off\n\t"
      "global_load_async_to_lds_b128 %0, %1, off offset:16"
      :: "v"(dst), "v"(gsrc)
      : "memory");
}

static __device__ __forceinline__ void wait_async0() {
  asm volatile("s_wait_asynccnt 0" ::: "memory");
}

// ---- tile loaders (global -> LDS), 256 threads fill a 128x32 bf16 tile -----
static __device__ __forceinline__ void load_a_async(__bf16 (*As)[LDSS],
    const __bf16* __restrict__ A, int blockM, int K, int kt, int tid) {
  const int row = tid >> 1, half = tid & 1;
  const __bf16* src = A + (size_t)(blockM + row) * (size_t)K + (size_t)kt * BK + half * 16;
  async_copy32(&As[row][half * 16], src);
}

// A-tile produced on the fly from f32 scores: P = exp(S - rowmax) * (1/rowsum)
static __device__ __forceinline__ void load_a_exp(__bf16 (*As)[LDSS],
    const float* __restrict__ S, float rowm, float rowr,
    int blockM, int K, int kt, int tid) {
  const int row = tid >> 1, half = tid & 1;
  const float* src = S + (size_t)(blockM + row) * (size_t)K + (size_t)kt * BK + half * 16;
  bf16x8 lo, hi;
  #pragma unroll
  for (int e = 0; e < 8; ++e) lo[e] = (__bf16)(__expf(src[e]     - rowm) * rowr);
  #pragma unroll
  for (int e = 0; e < 8; ++e) hi[e] = (__bf16)(__expf(src[8 + e] - rowm) * rowr);
  __builtin_prefetch(src + BK, 0, 1);
  *(bf16x8*)&As[row][half * 16]     = lo;
  *(bf16x8*)&As[row][half * 16 + 8] = hi;
}

// B from weight-style layout (N x K row-major): Bs[n][k] = B[n][k]  (async)
static __device__ __forceinline__ void load_b_async(__bf16 (*Bs)[LDSS],
    const __bf16* __restrict__ B, int blockN, int K, int kt, int tid) {
  const int row = tid >> 1, half = tid & 1;
  const __bf16* src = B + (size_t)(blockN + row) * (size_t)K + (size_t)kt * BK + half * 16;
  async_copy32(&Bs[row][half * 16], src);
}

// B from activation layout (K x N row-major): Bs[n][k] = B[k][n] (reg transpose)
static __device__ __forceinline__ void load_b_kn(__bf16 (*Bs)[LDSS],
    const __bf16* __restrict__ B, int blockN, int N, int kt, int tid) {
  const int kk = tid >> 3;            // 0..31
  const int ng = (tid & 7) * 16;      // 0..112
  const __bf16* src = B + ((size_t)kt * BK + kk) * (size_t)N + blockN + ng;
  bf16x8 v0 = *(const bf16x8*)src;
  bf16x8 v1 = *(const bf16x8*)(src + 8);
  #pragma unroll
  for (int e = 0; e < 8; ++e) Bs[ng + e][kk]     = v0[e];
  #pragma unroll
  for (int e = 0; e < 8; ++e) Bs[ng + 8 + e][kk] = v1[e];
}

// ---------------------------------------------------------------------------
// Universal bf16 WMMA GEMM: out = epilogue(A(MxK) * B(KxN))
// ---------------------------------------------------------------------------
template<int EPI, bool BT, bool AEXP>
__global__ __launch_bounds__(256)
void gemm_bf16_wmma(const __bf16* __restrict__ A, const float* __restrict__ Sexp,
                    const float* __restrict__ amx, const float* __restrict__ ars,
                    const __bf16* __restrict__ B,
                    const float* __restrict__ bias, const float* __restrict__ aux,
                    float* __restrict__ outF, __bf16* __restrict__ outB,
                    int M, int N, int K, float scale) {
  __shared__ __attribute__((aligned(16))) __bf16 As[2][BM][LDSS];
  __shared__ __attribute__((aligned(16))) __bf16 Bs[2][BN][LDSS];
  (void)M;
  constexpr bool USE_ASYNC = (!AEXP) || (!BT);  // any async staging in flight?

  const int tid  = threadIdx.x;
  const int lane = tid & 31;
  const int h    = lane >> 4;   // lane half (WMMA layout)
  const int ln   = lane & 15;
  const int wave = tid >> 5;
  const int mb   = (wave & 3) * 32;   // 4 waves along M
  const int nb   = (wave >> 2) * 64;  // 2 waves along N

  const int blockM = blockIdx.x * BM;
  const int blockN = blockIdx.y * BN;

  float rowm = 0.f, rowr = 0.f;
  if constexpr (AEXP) {
    const int rg = blockM + (tid >> 1);
    rowm = amx[rg]; rowr = ars[rg];
  }

  f32x8 acc[2][4];
  #pragma unroll
  for (int i = 0; i < 2; ++i)
    #pragma unroll
    for (int j = 0; j < 4; ++j)
      #pragma unroll
      for (int r = 0; r < 8; ++r) acc[i][j][r] = 0.0f;

  const int KT = K / BK;

  // prologue: stage tile 0
  if constexpr (AEXP) load_a_exp(As[0], Sexp, rowm, rowr, blockM, K, 0, tid);
  else                load_a_async(As[0], A, blockM, K, 0, tid);
  if constexpr (BT)   load_b_kn(Bs[0], B, blockN, N, 0, tid);
  else                load_b_async(Bs[0], B, blockN, K, 0, tid);
  if constexpr (USE_ASYNC) wait_async0();
  __syncthreads();

  for (int kt = 0; kt < KT; ++kt) {
    const int cur = kt & 1;
    if (kt + 1 < KT) {  // fire tile kt+1 into the other buffer (async, no VGPR staging)
      if constexpr (AEXP) load_a_exp(As[cur ^ 1], Sexp, rowm, rowr, blockM, K, kt + 1, tid);
      else                load_a_async(As[cur ^ 1], A, blockM, K, kt + 1, tid);
      if constexpr (BT)   load_b_kn(Bs[cur ^ 1], B, blockN, N, kt + 1, tid);
      else                load_b_async(Bs[cur ^ 1], B, blockN, K, kt + 1, tid);
    }

    // build fragments per CDNA5 16-bit WMMA VGPR layouts
    bf16x16 af[2], bfr[4];
    #pragma unroll
    for (int i = 0; i < 2; ++i) {
      const int r = mb + i * 16 + ln;                 // A row = M
      bf16x8 a0 = *(const bf16x8*)&As[cur][r][h * 8];        // K = h*8 .. +7
      bf16x8 a1 = *(const bf16x8*)&As[cur][r][16 + h * 8];   // K = 16+h*8 .. +7
      af[i] = cat8(a0, a1);
    }
    #pragma unroll
    for (int j = 0; j < 4; ++j) {
      const int c = nb + j * 16 + ln;                 // B col = N
      bf16x8 b0 = *(const bf16x8*)&Bs[cur][c][h * 8];        // K = h*8 .. +7
      bf16x8 b1 = *(const bf16x8*)&Bs[cur][c][16 + h * 8];   // K = 16+h*8 .. +7
      bfr[j] = cat8(b0, b1);
    }
    #pragma unroll
    for (int i = 0; i < 2; ++i)
      #pragma unroll
      for (int j = 0; j < 4; ++j)
        acc[i][j] = __builtin_amdgcn_wmma_f32_16x16x32_bf16(
            false, af[i], false, bfr[j], (short)0, acc[i][j], false, false);

    if constexpr (USE_ASYNC) wait_async0();  // tile kt+1 fully in LDS
    __syncthreads();
  }

  // epilogue: C/D layout -> element r is (m = h*8+r, n = ln) of each 16x16 tile
  #pragma unroll
  for (int i = 0; i < 2; ++i) {
    #pragma unroll
    for (int j = 0; j < 4; ++j) {
      const int col = blockN + nb + j * 16 + ln;
      #pragma unroll
      for (int r = 0; r < 8; ++r) {
        const int row = blockM + mb + i * 16 + h * 8 + r;
        const size_t o = (size_t)row * (size_t)N + col;
        const float v = acc[i][j][r];
        if constexpr (EPI == EPI_BIAS) {
          outF[o] = v + bias[col];
        } else if constexpr (EPI == EPI_SIGMOID) {
          outF[o] = 1.0f / (1.0f + __expf(-(v + bias[col])));
        } else if constexpr (EPI == EPI_ELU1) {   // elu(x)+1
          outB[o] = (__bf16)(v > 0.0f ? v + 1.0f : __expf(v));
        } else if constexpr (EPI == EPI_BF16) {
          outB[o] = (__bf16)v;
        } else if constexpr (EPI == EPI_SCALE) {
          outF[o] = v * scale;
        } else if constexpr (EPI == EPI_MULY) {   // att * y -> bf16
          outB[o] = (__bf16)(v * aux[o]);
        } else if constexpr (EPI == EPI_BIASRES) { // + bias + residual(xg)
          outF[o] = v + bias[col] + aux[o];
        }
      }
    }
  }
}

// ---------------------------------------------------------------------------
// helpers
// ---------------------------------------------------------------------------
__global__ __launch_bounds__(256)
void cvt_bf16(const float* __restrict__ in, __bf16* __restrict__ out, int n) {
  const int i = blockIdx.x * 256 + threadIdx.x;
  if (i < n) out[i] = (__bf16)in[i];
}

__global__ __launch_bounds__(256)
void conv3x3_gelu(const float* __restrict__ hsrc, const float* __restrict__ cw,
                  const float* __restrict__ cb, float* __restrict__ xg,
                  __bf16* __restrict__ xgb, int NR, int D) {
  const int idx = blockIdx.x * 256 + threadIdx.x;
  if (idx >= NR * D) return;
  const int n = idx / D, d = idx % D;
  float acc = cb[0];
  #pragma unroll
  for (int di = -1; di <= 1; ++di) {
    const int nn = n + di;
    if (nn < 0 || nn >= NR) continue;
    #pragma unroll
    for (int dj = -1; dj <= 1; ++dj) {
      const int dd = d + dj;
      if (dd < 0 || dd >= D) continue;
      acc += hsrc[(size_t)nn * D + dd] * cw[(di + 1) * 3 + (dj + 1)];
    }
  }
  const float g = 0.5f * acc * (1.0f + erff(acc * 0.70710678118f)); // exact GELU
  xg[idx]  = g;
  xgb[idx] = (__bf16)g;
}

// one block per row: rowmax + 1/sum(exp(.-max)); row stays hot in L2 for pass 2
__global__ __launch_bounds__(256)
void rowstats(const float* __restrict__ S, float* __restrict__ mx,
              float* __restrict__ rs, int K) {
  const int row = blockIdx.x;
  const float* p = S + (size_t)row * (size_t)K;
  const int tid = threadIdx.x, lane = tid & 31, wave = tid >> 5;
  __shared__ float red[8];
  float m = -3.402823466e38f;
  for (int i = tid; i < K; i += 256) m = fmaxf(m, p[i]);
  #pragma unroll
  for (int o = 16; o > 0; o >>= 1) m = fmaxf(m, __shfl_xor(m, o, 32));
  if (lane == 0) red[wave] = m;
  __syncthreads();
  float mf = red[0];
  #pragma unroll
  for (int w = 1; w < 8; ++w) mf = fmaxf(mf, red[w]);
  __syncthreads();
  float s = 0.0f;
  for (int i = tid; i < K; i += 256) s += __expf(p[i] - mf);
  #pragma unroll
  for (int o = 16; o > 0; o >>= 1) s += __shfl_xor(s, o, 32);
  if (lane == 0) red[wave] = s;
  __syncthreads();
  if (tid == 0) {
    float t = 0.0f;
    #pragma unroll
    for (int w = 0; w < 8; ++w) t += red[w];
    mx[row] = mf;
    rs[row] = 1.0f / t;
  }
}

// ---------------------------------------------------------------------------
extern "C" void kernel_launch(void* const* d_in, const int* in_sizes, int n_in,
                              void* d_out, int out_size, void* d_ws, size_t ws_size,
                              hipStream_t stream) {
  (void)in_sizes; (void)n_in; (void)out_size; (void)ws_size;
  const int   NR = 8192, D = 1024;
  const float NORM = 0.03125f;  // 1/sqrt(1024)

  const float* x  = (const float*)d_in[0];
  const float* W1 = (const float*)d_in[1];
  const float* b1 = (const float*)d_in[2];
  const float* W2 = (const float*)d_in[3];
  const float* b2 = (const float*)d_in[4];
  const float* cw = (const float*)d_in[5];
  const float* cb = (const float*)d_in[6];
  const float* Wq = (const float*)d_in[7];
  const float* Wk = (const float*)d_in[8];
  const float* Wv = (const float*)d_in[9];
  const float* Wo = (const float*)d_in[10];
  const float* bo = (const float*)d_in[11];
  float* out = (float*)d_out;

  const size_t ND = (size_t)NR * D, DD = (size_t)D * D, NN = (size_t)NR * NR;
  char* ws = (char*)d_ws;
  size_t off = 0;
  auto alloc = [&](size_t bytes) -> char* {
    char* p = ws + off;
    off += (bytes + 255) & ~(size_t)255;
    return p;
  };
  __bf16* xb  = (__bf16*)alloc(ND * 2);
  __bf16* w1b = (__bf16*)alloc(DD * 2);
  __bf16* w2b = (__bf16*)alloc(DD * 2);
  __bf16* wqb = (__bf16*)alloc(DD * 2);
  __bf16* wkb = (__bf16*)alloc(DD * 2);
  __bf16* wvb = (__bf16*)alloc(DD * 2);
  __bf16* wob = (__bf16*)alloc(DD * 2);
  float*  h   = (float*)alloc(ND * 4);
  float*  y   = (float*)alloc(ND * 4);
  float*  xg  = (float*)alloc(ND * 4);
  __bf16* xgb = (__bf16*)alloc(ND * 2);
  __bf16* qb  = (__bf16*)alloc(ND * 2);
  __bf16* kb  = (__bf16*)alloc(ND * 2);
  __bf16* vb  = (__bf16*)alloc(ND * 2);
  float*  S   = (float*)alloc(NN * 4);
  float*  mx  = (float*)alloc((size_t)NR * 4);
  float*  rs  = (float*)alloc((size_t)NR * 4);
  __bf16* gy  = (__bf16*)alloc(ND * 2);

  const dim3 blk(256);
  const dim3 gD(NR / BM, D / BN);    // (64, 8)
  const dim3 gS(NR / BM, NR / BN);   // (64, 64)

  // 0) operand conversion to bf16
  cvt_bf16<<<dim3((unsigned)((ND + 255) / 256)), blk, 0, stream>>>(x, xb, (int)ND);
  cvt_bf16<<<dim3((unsigned)((DD + 255) / 256)), blk, 0, stream>>>(W1, w1b, (int)DD);
  cvt_bf16<<<dim3((unsigned)((DD + 255) / 256)), blk, 0, stream>>>(W2, w2b, (int)DD);
  cvt_bf16<<<dim3((unsigned)((DD + 255) / 256)), blk, 0, stream>>>(Wq, wqb, (int)DD);
  cvt_bf16<<<dim3((unsigned)((DD + 255) / 256)), blk, 0, stream>>>(Wk, wkb, (int)DD);
  cvt_bf16<<<dim3((unsigned)((DD + 255) / 256)), blk, 0, stream>>>(Wv, wvb, (int)DD);
  cvt_bf16<<<dim3((unsigned)((DD + 255) / 256)), blk, 0, stream>>>(Wo, wob, (int)DD);

  // 1) h = x@W1^T + b1 ; 2) y = sigmoid(x@W2^T + b2)
  gemm_bf16_wmma<EPI_BIAS, false, false><<<gD, blk, 0, stream>>>(
      xb, nullptr, nullptr, nullptr, w1b, b1, nullptr, h, nullptr, NR, D, D, 1.0f);
  gemm_bf16_wmma<EPI_SIGMOID, false, false><<<gD, blk, 0, stream>>>(
      xb, nullptr, nullptr, nullptr, w2b, b2, nullptr, y, nullptr, NR, D, D, 1.0f);

  // 3) xg = gelu(conv3x3(h) + cb)
  conv3x3_gelu<<<dim3((unsigned)((ND + 255) / 256)), blk, 0, stream>>>(h, cw, cb, xg, xgb, NR, D);

  // 4) q = elu(xg@Wq^T)+1 ; k = elu(xg@Wk^T)+1 ; v = xg@Wv^T
  gemm_bf16_wmma<EPI_ELU1, false, false><<<gD, blk, 0, stream>>>(
      xgb, nullptr, nullptr, nullptr, wqb, nullptr, nullptr, nullptr, qb, NR, D, D, 1.0f);
  gemm_bf16_wmma<EPI_ELU1, false, false><<<gD, blk, 0, stream>>>(
      xgb, nullptr, nullptr, nullptr, wkb, nullptr, nullptr, nullptr, kb, NR, D, D, 1.0f);
  gemm_bf16_wmma<EPI_BF16, false, false><<<gD, blk, 0, stream>>>(
      xgb, nullptr, nullptr, nullptr, wvb, nullptr, nullptr, nullptr, vb, NR, D, D, 1.0f);

  // 5) S = (v @ k^T) * 1/sqrt(D)   (8192 x 8192)
  gemm_bf16_wmma<EPI_SCALE, false, false><<<gS, blk, 0, stream>>>(
      vb, nullptr, nullptr, nullptr, kb, nullptr, nullptr, S, nullptr, NR, NR, D, NORM);

  // 6) softmax row stats
  rowstats<<<dim3((unsigned)NR), blk, 0, stream>>>(S, mx, rs, NR);

  // 7) gy = (softmax(S) @ q) * y  -> bf16 (A built on the fly from S)
  gemm_bf16_wmma<EPI_MULY, true, true><<<gD, blk, 0, stream>>>(
      nullptr, S, mx, rs, qb, nullptr, y, nullptr, gy, NR, D, NR, 1.0f);

  // 8) out = gy @ Wo^T + bo + xg
  gemm_bf16_wmma<EPI_BIASRES, false, false><<<gD, blk, 0, stream>>>(
      gy, nullptr, nullptr, nullptr, wob, bo, xg, out, nullptr, NR, D, D, 1.0f);
}
